// ParallelRNNBlock_4801773437172
// MI455X (gfx1250) — matrix operand order repaired and checked by reading, not compiled, not run on hardware
//
#include <hip/hip_runtime.h>
#include <hip/hip_bf16.h>

typedef _Float16 v16h __attribute__((ext_vector_type(16)));
typedef float    v8f  __attribute__((ext_vector_type(8)));

#define BATCH   8
#define SEQL    512
#define IN_DIM  768
#define FD      512
#define SD      512
#define HR      256

#define LDSW 36   // 32 k-floats + 4 pad floats per row (144B: 16B aligned, bank-spread)

// Convert two 8-float LDS chunks into one 16-half WMMA fragment register set.
__device__ __forceinline__ v16h cvt16(const float* __restrict__ a,
                                      const float* __restrict__ b) {
  v16h f;
#pragma unroll
  for (int j = 0; j < 8; ++j) f[j] = (_Float16)a[j];
#pragma unroll
  for (int j = 0; j < 8; ++j) f[8 + j] = (_Float16)b[j];
  return f;
}

// C/D 16x16 f32: VGPR r, lanes 0-15 -> (M=r, N=lane); lanes 16-31 -> (M=r+8, N=lane-16)
__device__ __forceinline__ void store_acc(float* __restrict__ Cb, int ldc,
                                          int tm, int tn, int lane, v8f acc,
                                          const float* __restrict__ bias,
                                          int biasMode, int relu) {
  int n = tn + (lane & 15);
  int mBase = tm + ((lane >> 4) << 3);
  float bn = 0.f;
  if (biasMode == 1) bn = bias[n];
#pragma unroll
  for (int r = 0; r < 8; ++r) {
    int m = mBase + r;
    float v = acc[r];
    if (biasMode == 2) v += bias[m]; else v += bn;
    if (relu) v = v > 0.f ? v : 0.f;
    Cb[(size_t)m * ldc + n] = v;
  }
}

// Issue one async global->LDS 16B copy (per-lane dsaddr), plus the +16B chunk.
__device__ __forceinline__ void async_copy32B(const float* __restrict__ gptr,
                                              float* __restrict__ lptr) {
  uint32_t la = (uint32_t)(uintptr_t)lptr;          // flat LDS addr low 32 = LDS offset
  uint64_t ga = (uint64_t)(uintptr_t)gptr;
  asm volatile("global_load_async_to_lds_b128 %0, %1, off"
               :: "v"(la), "v"(ga) : "memory");
  asm volatile("global_load_async_to_lds_b128 %0, %1, off offset:16"
               :: "v"(la), "v"(ga) : "memory");
}

// ---------------------------------------------------------------------------
// WMMA GEMM, LDS-staged. NT: C[m,n] = sum_k A[m,k]*B[n,k]  (A:[M,K], B:[N,K])
//                        NN: C[m,n] = sum_k A[m,k]*B[k,n]  (B:[K,N] N-contig)
// Block: 512 threads = 16 waves, C supertile 128x128 (waves 4x4, 32x32 each).
// A (and NT-B) tiles staged fp32 via global_load_async_to_lds_b128; NN-B tile
// transposed into LDS with per-element stores once per block (not per wave).
// Fragments per ISA 7.12.2:
//   A f16 16x32: lane l row m=l&15; half j -> K = 16*(j>>3) + 8*(l>>4) + (j&7)
//   B f16 32x16: lane l col n=l&15; half j -> K = 16*(j>>3) + 8*(l>>4) + (j&7)
// biasMode: 0 none, 1 bias[n], 2 bias[m]. Batched via blockIdx.z + strides.
// ---------------------------------------------------------------------------
template <bool NN>
__global__ void __launch_bounds__(512)
gemm_wmma(const float* __restrict__ A, const float* __restrict__ B,
          float* __restrict__ C, const float* __restrict__ bias,
          int M, int N, int K,
          long strideA, long strideB, long strideC,
          int biasMode, int relu) {
  __shared__ float As[128 * LDSW];
  __shared__ float Bs[128 * LDSW];

  int tid  = threadIdx.x;
  int lane = tid & 31;
  int wid  = tid >> 5;
  int wm   = wid >> 2, wn = wid & 3;

  int tilesN = N >> 7;
  int tm0 = (blockIdx.x / tilesN) << 7;
  int tn0 = (blockIdx.x % tilesN) << 7;

  const float* Ab = A + strideA * blockIdx.z;
  const float* Bb = B + strideB * blockIdx.z;
  float*       Cb = C + strideC * blockIdx.z;

  int lr = lane & 15, lh = lane >> 4;
  int ar0 = wm * 32 + lr, ar1 = ar0 + 16;      // block-local A rows for this wave
  int bc0 = wn * 32 + lr, bc1 = bc0 + 16;      // block-local B cols for this wave

  // staging coordinates
  int srow = tid >> 2, skq = (tid & 3) << 3;   // A / NT-B: row 0..127, 8 floats
  int skk = tid >> 4, sn0 = (tid & 15) << 3;   // NN-B: k 0..31, 8 n-values

  v8f acc00 = {}, acc01 = {}, acc10 = {}, acc11 = {};

  for (int k0 = 0; k0 < K; k0 += 32) {
    __syncthreads();   // previous iteration's LDS reads complete

    // ---- stage A tile [128 x 32] (async DMA to LDS, fp32) ----
    async_copy32B(Ab + (size_t)(tm0 + srow) * K + k0 + skq,
                  &As[srow * LDSW + skq]);

    if (NN) {
      // ---- stage B tile: Bs[n][k] = B[k0+k][tn0+n] (transposing scatter) ----
      const float* g = Bb + (size_t)(k0 + skk) * N + tn0 + sn0;
      float v0 = g[0], v1 = g[1], v2 = g[2], v3 = g[3];
      float v4 = g[4], v5 = g[5], v6 = g[6], v7 = g[7];
      float* d = &Bs[sn0 * LDSW + skk];
      d[0*LDSW] = v0; d[1*LDSW] = v1; d[2*LDSW] = v2; d[3*LDSW] = v3;
      d[4*LDSW] = v4; d[5*LDSW] = v5; d[6*LDSW] = v6; d[7*LDSW] = v7;
    } else {
      // ---- stage B tile [128 x 32] row-major (async DMA to LDS) ----
      async_copy32B(Bb + (size_t)(tn0 + srow) * K + k0 + skq,
                    &Bs[srow * LDSW + skq]);
    }

    asm volatile("s_wait_asynccnt 0x0" ::: "memory");
    __syncthreads();   // staged tiles visible to all waves

    // ---- fragments from LDS + 4 WMMAs (2x2 register tiling) ----
    v16h a0 = cvt16(&As[ar0 * LDSW + 8 * lh], &As[ar0 * LDSW + 16 + 8 * lh]);
    v16h a1 = cvt16(&As[ar1 * LDSW + 8 * lh], &As[ar1 * LDSW + 16 + 8 * lh]);
    v16h b0 = cvt16(&Bs[bc0 * LDSW + 8 * lh], &Bs[bc0 * LDSW + 16 + 8 * lh]);
    v16h b1 = cvt16(&Bs[bc1 * LDSW + 8 * lh], &Bs[bc1 * LDSW + 16 + 8 * lh]);

    acc00 = __builtin_amdgcn_wmma_f32_16x16x32_f16(false, a0, false, b0, (short)0, acc00, false, false);
    acc01 = __builtin_amdgcn_wmma_f32_16x16x32_f16(false, a0, false, b1, (short)0, acc01, false, false);
    acc10 = __builtin_amdgcn_wmma_f32_16x16x32_f16(false, a1, false, b0, (short)0, acc10, false, false);
    acc11 = __builtin_amdgcn_wmma_f32_16x16x32_f16(false, a1, false, b1, (short)0, acc11, false, false);
  }

  int gm = tm0 + wm * 32, gn = tn0 + wn * 32;
  store_acc(Cb, N, gm,      gn,      lane, acc00, bias, biasMode, relu);
  store_acc(Cb, N, gm,      gn + 16, lane, acc01, bias, biasMode, relu);
  store_acc(Cb, N, gm + 16, gn,      lane, acc10, bias, biasMode, relu);
  store_acc(Cb, N, gm + 16, gn + 16, lane, acc11, bias, biasMode, relu);
}

// x + positional encoding
__global__ void add_pe_kernel(const float* __restrict__ x, const float* __restrict__ pe,
                              float* __restrict__ out, int total, int seq, int d) {
  int i = blockIdx.x * blockDim.x + threadIdx.x;
  if (i >= total) return;
  int k = i % d;
  int s = (i / d) % seq;
  out[i] = x[i] + pe[s * d + k];
}

// Linear recurrence: c_{-1}=1; c_t = lam*c_{t-1} + Bx_t  (== masked lam-power einsum
// + lam^{t+1} decay of the reference).  Bx laid out [B, H, L] (contiguous per channel),
// output [B, L, 2H] = concat(real, imag).
__global__ void rnn_scan_kernel(const float* __restrict__ Bx,
                                const float* __restrict__ nu,
                                const float* __restrict__ theta,
                                float* __restrict__ out, int Bn, int H, int L) {
  int i = blockIdx.x * blockDim.x + threadIdx.x;
  if (i >= Bn * H) return;
  int h = i % H, b = i / H;
  float mag = expf(-expf(nu[h]));
  float lr = mag * cosf(theta[h]);
  float li = mag * sinf(theta[h]);
  const float* src = Bx + (size_t)i * L;
  float* ob = out + (size_t)b * L * 2 * H;
  float cr = 1.f, ci = 0.f;
  for (int t = 0; t < L; ++t) {
    float nr = lr * cr - li * ci + src[t];
    float ni = lr * ci + li * cr;
    ob[(size_t)t * 2 * H + h]     = nr;
    ob[(size_t)t * 2 * H + H + h] = ni;
    cr = nr; ci = ni;
  }
}

// [b, R, C] -> [b, C, R]
__global__ void transpose_kernel(const float* __restrict__ in, float* __restrict__ out,
                                 int R, int C) {
  __shared__ float tile[32][33];
  int b = blockIdx.z;
  int c0 = blockIdx.x * 32, r0 = blockIdx.y * 32;
  const float* ib = in + (size_t)b * R * C;
  float* ob = out + (size_t)b * R * C;
  int tx = threadIdx.x, ty = threadIdx.y;
#pragma unroll
  for (int j = 0; j < 32; j += 8)
    tile[ty + j][tx] = ib[(size_t)(r0 + ty + j) * C + (c0 + tx)];
  __syncthreads();
#pragma unroll
  for (int j = 0; j < 32; j += 8)
    ob[(size_t)(c0 + ty + j) * R + (r0 + tx)] = tile[tx][ty + j];
}

// y[b,s] = sum_g w[g] * Y[b,g,s] + b0    (Y stored [B, G, S])
__global__ void out_head_kernel(const float* __restrict__ Y, const float* __restrict__ w,
                                const float* __restrict__ b0, float* __restrict__ y,
                                int total, int G, int S) {
  int i = blockIdx.x * blockDim.x + threadIdx.x;
  if (i >= total) return;
  int b = i / S, s = i % S;
  const float* Yb = Y + (size_t)b * G * S + s;
  float acc = b0[0];
  for (int g = 0; g < G; ++g) acc += w[g] * Yb[(size_t)g * S];
  y[i] = acc;
}

extern "C" void kernel_launch(void* const* d_in, const int* in_sizes, int n_in,
                              void* d_out, int out_size, void* d_ws, size_t ws_size,
                              hipStream_t stream) {
  const float* x     = (const float*)d_in[0];
  const float* pe    = (const float*)d_in[1];
  const float* w1w   = (const float*)d_in[2];
  const float* w1b   = (const float*)d_in[3];
  const float* w2w   = (const float*)d_in[4];
  const float* w2b   = (const float*)d_in[5];
  const float* r1B   = (const float*)d_in[6];
  const float* r1nu  = (const float*)d_in[7];
  const float* r1th  = (const float*)d_in[8];
  const float* l1w   = (const float*)d_in[9];
  const float* l1b   = (const float*)d_in[10];
  const float* l2w   = (const float*)d_in[11];
  const float* l2b   = (const float*)d_in[12];
  const float* r2B   = (const float*)d_in[13];
  const float* r2nu  = (const float*)d_in[14];
  const float* r2th  = (const float*)d_in[15];
  const float* l3w   = (const float*)d_in[16];
  const float* l3b   = (const float*)d_in[17];
  const float* l4w   = (const float*)d_in[18];
  const float* l4b   = (const float*)d_in[19];
  const float* o1w   = (const float*)d_in[20];
  const float* o1b   = (const float*)d_in[21];
  const float* o2w   = (const float*)d_in[22];
  const float* o2b   = (const float*)d_in[23];

  float* out = (float*)d_out;                 // [4096] y ++ [8*512*512] h
  float* ws  = (float*)d_ws;

  const long ACT = 4096L * 512;               // 2,097,152
  const long T0  = 4096L * 768;               // 3,145,728
  float* t0   = ws;                           // o0: t0 -> R1 -> R3
  float* H1   = ws + T0;                      // o1: H1 -> R2 -> R4
  float* Hh   = ws + T0 + ACT;                // o2: H  -> out2
  float* o3   = ws + T0 + 2 * ACT;            // o3: out1 -> Y1
  float* o4   = ws + T0 + 3 * ACT;            // o4: Bx1 -> Bx2
  float* R1 = t0, *R3 = t0;
  float* R2 = H1, *R4 = H1;
  float* out2 = Hh;
  float* out1 = o3, *Y1 = o3;
  float* Bx1 = o4, *Bx2 = o4;

  const long AS = 512L * 512;                 // per-batch activation stride
  const long BS = 256L * 512;                 // per-batch Bx stride

  // 1. t0 = x + pe                                       [8,512,768]
  add_pe_kernel<<<dim3((int)((T0 + 255) / 256)), 256, 0, stream>>>(
      x, pe, t0, (int)T0, SEQL, IN_DIM);

  // 2. H1 = relu(t0 @ w1^T + b1)                         [4096,512]  (NT)
  gemm_wmma<false><<<dim3(128, 1, 1), 512, 0, stream>>>(
      t0, w1w, H1, w1b, 4096, FD, IN_DIM, 0, 0, 0, 1, 1);

  // 3. H[b,s,f] = W2 @ H1[b] + b2[s]                     [b,512,512] (NN, bias on M)
  gemm_wmma<true><<<dim3(16, 1, BATCH), 512, 0, stream>>>(
      w2w, H1, Hh, w2b, SD, FD, SEQL, 0, AS, AS, 2, 0);

  // 4. Bx1[b,h,s] = B1 @ H[b]^T                          [b,256,512] (NT)
  gemm_wmma<false><<<dim3(8, 1, BATCH), 512, 0, stream>>>(
      r1B, Hh, Bx1, nullptr, HR, SD, FD, 0, AS, BS, 0, 0);

  // 5. out1[b,s,2H] = scan(lam1, Bx1)
  rnn_scan_kernel<<<dim3(8), 256, 0, stream>>>(Bx1, r1nu, r1th, out1, BATCH, HR, SEQL);

  // 6. R1 = relu(out1 @ l1^T + l1b)                      [4096,512] (NT)
  gemm_wmma<false><<<dim3(128, 1, 1), 512, 0, stream>>>(
      out1, l1w, R1, l1b, 4096, FD, 2 * HR, 0, 0, 0, 1, 1);

  // 7. R2 = R1 @ l2^T + l2b                              [4096,512] (NT)
  gemm_wmma<false><<<dim3(128, 1, 1), 512, 0, stream>>>(
      R1, l2w, R2, l2b, 4096, FD, FD, 0, 0, 0, 1, 0);

  // 8. Bx2[b,h,f] = B2 @ R2[b]                           [b,256,512] (NN)
  gemm_wmma<true><<<dim3(8, 1, BATCH), 512, 0, stream>>>(
      r2B, R2, Bx2, nullptr, HR, FD, SD, 0, AS, BS, 0, 0);

  // 9. out2[b,f,2H] = scan(lam2, Bx2)
  rnn_scan_kernel<<<dim3(8), 256, 0, stream>>>(Bx2, r2nu, r2th, out2, BATCH, HR, FD);

  // 10. R3 = relu(out2 @ l3^T + l3b)                     [4096,512] (NT)
  gemm_wmma<false><<<dim3(128, 1, 1), 512, 0, stream>>>(
      out2, l3w, R3, l3b, 4096, SD, 2 * HR, 0, 0, 0, 1, 1);

  // 11. R4 = R3 @ l4^T + l4b                             [b,f,s] (NT)
  gemm_wmma<false><<<dim3(128, 1, 1), 512, 0, stream>>>(
      R3, l4w, R4, l4b, 4096, SD, SD, 0, 0, 0, 1, 0);

  // 12. h_out[b,s,f] = R4[b,f,s]  -> second tuple output
  transpose_kernel<<<dim3(16, 16, BATCH), dim3(32, 8), 0, stream>>>(
      R4, out + 4096, FD, SD);

  // 13. Y1[b,g,s] = relu(O1 @ R4[b] + o1b[g])            [b,512,512] (NN, bias on M)
  gemm_wmma<true><<<dim3(16, 1, BATCH), 512, 0, stream>>>(
      o1w, R4, Y1, o1b, FD, SD, FD, 0, AS, AS, 2, 1);

  // 14. y[b,s] = sum_g o2w[g]*Y1[b,g,s] + o2b            -> first tuple output
  out_head_kernel<<<dim3(16), 256, 0, stream>>>(Y1, o2w, o2b, out, BATCH * SD, FD, SD);
}
